// ElectraGenerator_1803886265508
// MI455X (gfx1250) — matrix operand order repaired and checked by reading, not compiled, not run on hardware
//
#include <hip/hip_runtime.h>
#include <math.h>

#define NTOK   1024      // B*S = 2*512
#define DIMM   256
#define FFDIM  1024
#define VOCABN 32000
#define NLAYER 4

typedef __bf16 bf16;
typedef bf16  v16bf __attribute__((ext_vector_type(16)));
typedef bf16  v8bf  __attribute__((ext_vector_type(8)));
typedef bf16  v4bf  __attribute__((ext_vector_type(4)));
typedef float v8f   __attribute__((ext_vector_type(8)));
typedef float v4f   __attribute__((ext_vector_type(4)));

// f32 -> bf16 round-to-nearest-even via bit trick
__device__ __forceinline__ bf16 f2bf(float f) {
  unsigned u = __builtin_bit_cast(unsigned, f);
  u += 0x7FFFu + ((u >> 16) & 1u);
  unsigned short h = (unsigned short)(u >> 16);
  return __builtin_bit_cast(bf16, h);
}

// ---------------------------------------------------------------------------
// One-time weight conversion f32 -> bf16 (4 elements / thread, 8B stores)
// ---------------------------------------------------------------------------
__global__ void cvt_bf16v(const float* __restrict__ s, bf16* __restrict__ d, int n4) {
  int i = blockIdx.x * blockDim.x + threadIdx.x;
  if (i < n4) {
    v4f v = ((const v4f*)s)[i];
    union { v4bf b; unsigned long long u; } o;
#pragma unroll
    for (int c = 0; c < 4; ++c) o.b[c] = f2bf(v[c]);
    ((unsigned long long*)d)[i] = o.u;
  }
}

// ---------------------------------------------------------------------------
// WMMA GEMM core.  C[M,Nout] = A_bf16[M,Kt] * W[Nout,Kt]^T  (KAN: base phase
// K=in then spline phase K=8*in, accumulators carried).  Wave32, each wave
// owns a 16x64 C tile via v_wmma_f32_16x16x32_bf16.
// A frag per lane (m=lane&15, hi=lane>>4): 16 consecutive K halfs at k0+16hi
// B frag per lane (n fixed): 16 consecutive K halfs at k0+16hi.
// WBF=true: weights pre-converted bf16 -> pure b128 loads, zero VALU in loop.
// WBF=false: fallback, convert f32 weights in-register.
// ---------------------------------------------------------------------------
template <bool WBF>
__device__ __forceinline__ void gemm_phase_t(const bf16* __restrict__ A, int lda,
                                             const void* __restrict__ Wv, int ldw,
                                             int K, int m, int hi, int nlane,
                                             v8f acc[4]) {
  const bf16* arow = A + (size_t)m * lda + 16 * hi;
  for (int k0 = 0; k0 < K; k0 += 32) {
    union { v16bf v; v8bf h[2]; } af;
    const v8bf* pa = (const v8bf*)(arow + k0);
    af.h[0] = pa[0];   // K = k0+16hi .. +7
    af.h[1] = pa[1];   // K = k0+16hi+8 .. +15
#pragma unroll
    for (int j = 0; j < 4; ++j) {
      union { v16bf v; v8bf h[2]; bf16 e[16]; } bfrag;
      if constexpr (WBF) {
        const bf16* wrow = (const bf16*)Wv + (size_t)(nlane + 16 * j) * ldw + k0 + 16 * hi;
        __builtin_prefetch(wrow + 64, 0, 1);         // K-block after next of this row
        const v8bf* pw = (const v8bf*)wrow;
        bfrag.h[0] = pw[0];
        bfrag.h[1] = pw[1];
      } else {
        const float* wrow = (const float*)Wv + (size_t)(nlane + 16 * j) * ldw + k0 + 16 * hi;
        __builtin_prefetch(wrow + 32, 0, 1);
        const v4f* pw = (const v4f*)wrow;
#pragma unroll
        for (int q4 = 0; q4 < 4; ++q4) {
          v4f wv = pw[q4];
#pragma unroll
          for (int c = 0; c < 4; ++c) bfrag.e[q4 * 4 + c] = f2bf(wv[c]);
        }
      }
      acc[j] = __builtin_amdgcn_wmma_f32_16x16x32_bf16(
          false, af.v, false, bfrag.v, (short)0, acc[j], false, false);
    }
  }
}

template <bool WBF>
__global__ void __launch_bounds__(256) kan_gemm_t(
    const bf16* __restrict__ A, int lda,            // [M, 9*in] bf16
    const void* __restrict__ Wb,                    // [Nout, in]
    const void* __restrict__ Ws,                    // [Nout, 8*in]
    int in, int M, int Nout, float* __restrict__ C) {
  int wid    = blockIdx.x * (blockDim.x >> 5) + (threadIdx.x >> 5);
  int lane   = threadIdx.x & 31;
  int tilesN = Nout >> 6;
  int tm = wid / tilesN;
  int tn = wid - tm * tilesN;
  if (tm * 16 >= M) return;                         // wave-uniform (never taken)
  int m  = tm * 16 + (lane & 15);
  int hi = lane >> 4;
  int nl = tn * 64 + (lane & 15);
  v8f acc[4] = {};
  gemm_phase_t<WBF>(A,      lda, Wb, in,     in,     m, hi, nl, acc);
  gemm_phase_t<WBF>(A + in, lda, Ws, 8 * in, 8 * in, m, hi, nl, acc);
#pragma unroll
  for (int j = 0; j < 4; ++j) {
    int n = nl + 16 * j;
#pragma unroll
    for (int r = 0; r < 8; ++r) {
      int row = tm * 16 + r + 8 * hi;               // C/D layout: VGPR r -> M=r+8*hi
      C[(size_t)row * Nout + n] = acc[j][r];
    }
  }
}

// ---------------------------------------------------------------------------
// KAN activation prep: A = [ silu(x) | b_splines(x) flattened (i,c)->i*8+c ]
// Uniform extended grid: grid(j) = (j-3)*0.4 - 1, order 3, 8 coefficients.
// ---------------------------------------------------------------------------
__global__ void kan_prep(const float* __restrict__ X, bf16* __restrict__ A,
                         int in, int lda, int ntok) {
  int total = ntok * in;
  for (int idx = blockIdx.x * blockDim.x + threadIdx.x; idx < total;
       idx += gridDim.x * blockDim.x) {
    int n = idx / in, i = idx - n * in;
    float x = X[idx];
    float s = x / (1.f + __expf(-x));               // silu
    A[(size_t)n * lda + i] = f2bf(s);
    const float h = 0.4f;
    float bb[11];
#pragma unroll
    for (int t = 0; t < 11; ++t) {
      float g0 = (float)(t - 3) * h - 1.f;
      float g1 = (float)(t - 2) * h - 1.f;
      bb[t] = (x >= g0 && x < g1) ? 1.f : 0.f;
    }
#pragma unroll
    for (int k = 1; k <= 3; ++k) {
      float inv = 1.f / ((float)k * h);
#pragma unroll
      for (int t = 0; t < 10; ++t) {
        if (t + k < 11) {
          float gl = (float)(t - 3) * h - 1.f;
          float gr = (float)(t + k - 2) * h - 1.f;  // grid(t+k+1)
          bb[t] = (x - gl) * inv * bb[t] + (gr - x) * inv * bb[t + 1];
        }
      }
    }
    bf16* dst = A + (size_t)n * lda + in + (size_t)i * 8;
#pragma unroll
    for (int c = 0; c < 8; ++c) dst[c] = f2bf(bb[c]);
  }
}

// ---------------------------------------------------------------------------
__global__ void embed_gather(const int* __restrict__ ids,
                             const float* __restrict__ we,
                             const float* __restrict__ tt,
                             const float* __restrict__ pe,
                             float* __restrict__ X) {
  int t = blockIdx.x, d = threadIdx.x;
  int id = ids[t];
  X[(size_t)t * DIMM + d] = we[(size_t)id * DIMM + d] + tt[DIMM + d] + pe[DIMM + d];
}

// (optional add) + LayerNorm over DIM=256, optional affine. One block per row.
__global__ void add_ln(const float* __restrict__ Aa, const float* __restrict__ Bb,
                       const float* __restrict__ g, const float* __restrict__ bp,
                       float* __restrict__ O, float eps) {
  __shared__ float red[256];
  int row = blockIdx.x, d = threadIdx.x;
  float v = Aa[(size_t)row * DIMM + d];
  if (Bb) v += Bb[(size_t)row * DIMM + d];
  red[d] = v; __syncthreads();
  for (int s = 128; s > 0; s >>= 1) { if (d < s) red[d] += red[d + s]; __syncthreads(); }
  float mu = red[0] * (1.f / 256.f); __syncthreads();
  float c = v - mu;
  red[d] = c * c; __syncthreads();
  for (int s = 128; s > 0; s >>= 1) { if (d < s) red[d] += red[d + s]; __syncthreads(); }
  float var = red[0] * (1.f / 256.f);
  float y = c * rsqrtf(var + eps);
  if (g) y = y * g[d] + bp[d];
  O[(size_t)row * DIMM + d] = y;
}

// Attention: 4 heads, hd=64, S=512, scale=1/8. One wave per (b,h,q) row.
__global__ void __launch_bounds__(128) attention4(const float* __restrict__ Q,
                                                  const float* __restrict__ Kk,
                                                  const float* __restrict__ V,
                                                  float* __restrict__ CTX) {
  __shared__ float probs[4][512];
  int wave = threadIdx.x >> 5, lane = threadIdx.x & 31;
  int gw = blockIdx.x * 4 + wave;          // 0..4095
  int q  = gw & 511;
  int bh = gw >> 9;                        // 0..7
  int b = bh >> 2, hh = bh & 3;
  const float* qrow = Q + ((size_t)(b * 512 + q)) * DIMM + hh * 64;
  float sc[16];
  float mx = -1e30f;
#pragma unroll 4
  for (int kk = 0; kk < 16; ++kk) {
    int key = lane + kk * 32;
    const float* krow = Kk + ((size_t)(b * 512 + key)) * DIMM + hh * 64;
    float s = 0.f;
    for (int d = 0; d < 64; ++d) s = fmaf(qrow[d], krow[d], s);
    s *= 0.125f;
    sc[kk] = s; mx = fmaxf(mx, s);
  }
  for (int off = 16; off > 0; off >>= 1) mx = fmaxf(mx, __shfl_xor(mx, off));
  float sum = 0.f;
#pragma unroll
  for (int kk = 0; kk < 16; ++kk) { float e = __expf(sc[kk] - mx); sc[kk] = e; sum += e; }
  for (int off = 16; off > 0; off >>= 1) sum += __shfl_xor(sum, off);
  float inv = 1.f / sum;
#pragma unroll
  for (int kk = 0; kk < 16; ++kk) probs[wave][lane + kk * 32] = sc[kk] * inv;
  __syncthreads();
  float a0 = 0.f, a1 = 0.f;
  for (int key = 0; key < 512; ++key) {
    float p = probs[wave][key];
    const float* vrow = V + ((size_t)(b * 512 + key)) * DIMM + hh * 64;
    a0 = fmaf(p, vrow[lane], a0);
    a1 = fmaf(p, vrow[lane + 32], a1);
  }
  float* crow = CTX + ((size_t)(b * 512 + q)) * DIMM + hh * 64;
  crow[lane] = a0; crow[lane + 32] = a1;
}

__global__ void relu_k(float* X, int n) {
  for (int i = blockIdx.x * blockDim.x + threadIdx.x; i < n; i += gridDim.x * blockDim.x)
    X[i] = fmaxf(X[i], 0.f);
}
__global__ void gelu_k(float* X, int n) {
  for (int i = blockIdx.x * blockDim.x + threadIdx.x; i < n; i += gridDim.x * blockDim.x) {
    float x = X[i];
    X[i] = 0.5f * x * (1.f + erff(x * 0.70710678118654752f));
  }
}

// ---------------------------------------------------------------------------
extern "C" void kernel_launch(void* const* d_in, const int* in_sizes, int n_in,
                              void* d_out, int out_size, void* d_ws, size_t ws_size,
                              hipStream_t stream) {
  (void)in_sizes; (void)n_in; (void)out_size;
  const int*   input_ids = (const int*)  d_in[0];
  const float* word_emb  = (const float*)d_in[1];
  const float* tt_emb    = (const float*)d_in[2];
  const float* pos_emb   = (const float*)d_in[3];
  const float* emb_g     = (const float*)d_in[4];
  const float* emb_b     = (const float*)d_in[5];
  const float* ln1_g = (const float*)d_in[18];    const float* ln1_b = (const float*)d_in[19];
  const float* ln2_g = (const float*)d_in[20];    const float* ln2_b = (const float*)d_in[21];
  float* out = (float*)d_out;

  char* ws = (char*)d_ws;
  size_t off = 0;
  auto alloc = [&](size_t bytes) -> void* {
    void* p = ws + off; off += (bytes + 255) & ~(size_t)255; return p;
  };
  float* X   = (float*)alloc((size_t)NTOK * DIMM * 4);
  float* Qb  = (float*)alloc((size_t)NTOK * DIMM * 4);
  float* Kb  = (float*)alloc((size_t)NTOK * DIMM * 4);
  float* Vb  = (float*)alloc((size_t)NTOK * DIMM * 4);
  float* CTX = (float*)alloc((size_t)NTOK * DIMM * 4);
  float* T0  = (float*)alloc((size_t)NTOK * FFDIM * 4);     // widest intermediate
  float* T1  = (float*)alloc((size_t)NTOK * DIMM * 4);
  bf16*  AP  = (bf16*) alloc((size_t)NTOK * 9 * FFDIM * 2); // [1024, 9216] bf16

  const int LB = 256;

  // --- bf16 weight caches, allocated in FLOP-priority order while they fit.
  // (ws_size is constant across calls -> deterministic work.)
  const bf16* bfp[26] = {};
  struct PW { int bi, si; size_t bn, sn; };
  const size_t QN = (size_t)NLAYER * DIMM * DIMM;   // q/k/v/o base elems
  const size_t FN = (size_t)NLAYER * FFDIM * DIMM;  // f1/f2 base elems
  const PW prio[8] = {
    {24, 25, (size_t)VOCABN * DIMM, (size_t)VOCABN * DIMM * 8},   // ho (dominant)
    {14, 15, FN, FN * 8},                                         // f1
    {16, 17, FN, FN * 8},                                         // f2
    {6, 7, QN, QN * 8}, {8, 9, QN, QN * 8},                       // q, k
    {10, 11, QN, QN * 8}, {12, 13, QN, QN * 8},                   // v, o
    {22, 23, (size_t)DIMM * DIMM, (size_t)DIMM * DIMM * 8},       // hk
  };
  auto cvt = [&](int idx, size_t n) {
    bf16* dptr = (bf16*)alloc(n * 2);
    int n4 = (int)(n >> 2);
    cvt_bf16v<<<(n4 + LB - 1) / LB, LB, 0, stream>>>((const float*)d_in[idx], dptr, n4);
    bfp[idx] = dptr;
  };
  for (int p = 0; p < 8; ++p) {
    size_t need = ((prio[p].bn * 2 + 255) & ~(size_t)255) +
                  ((prio[p].sn * 2 + 255) & ~(size_t)255);
    if (off + need <= ws_size) { cvt(prio[p].bi, prio[p].bn); cvt(prio[p].si, prio[p].sn); }
  }

  // GEMM dispatch: bf16 weight cache if present, else f32-convert fallback.
  auto gemm = [&](const bf16* Aap, int lda, int bi, int si,
                  size_t offB, size_t offS, int in, int Nout, float* C) {
    int blocks = ((NTOK / 16) * (Nout / 64)) / 8;
    if (bfp[bi] && bfp[si])
      kan_gemm_t<true><<<blocks, 256, 0, stream>>>(
          Aap, lda, bfp[bi] + offB, bfp[si] + offS, in, NTOK, Nout, C);
    else
      kan_gemm_t<false><<<blocks, 256, 0, stream>>>(
          Aap, lda, (const float*)d_in[bi] + offB, (const float*)d_in[si] + offS,
          in, NTOK, Nout, C);
  };

  // Embedding + LN (eps 1e-12, affine)
  embed_gather<<<NTOK, 256, 0, stream>>>(input_ids, word_emb, tt_emb, pos_emb, X);
  add_ln<<<NTOK, 256, 0, stream>>>(X, nullptr, emb_g, emb_b, X, 1e-12f);

  for (int l = 0; l < NLAYER; ++l) {
    size_t wO  = (size_t)l * DIMM * DIMM;        // q/k/v/o base layer stride
    size_t wOs = wO * 8;
    size_t wF  = (size_t)l * FFDIM * DIMM;       // f1/f2 base layer stride
    size_t wFs = wF * 8;

    // Q,K,V projections (share one prep of X)
    kan_prep<<<NTOK * DIMM / LB, LB, 0, stream>>>(X, AP, DIMM, 9 * DIMM, NTOK);
    gemm(AP, 9 * DIMM, 6, 7, wO, wOs, DIMM, DIMM, Qb);
    gemm(AP, 9 * DIMM, 8, 9, wO, wOs, DIMM, DIMM, Kb);
    gemm(AP, 9 * DIMM, 10, 11, wO, wOs, DIMM, DIMM, Vb);

    attention4<<<1024, 128, 0, stream>>>(Qb, Kb, Vb, CTX);

    // Output projection + residual LN1
    kan_prep<<<NTOK * DIMM / LB, LB, 0, stream>>>(CTX, AP, DIMM, 9 * DIMM, NTOK);
    gemm(AP, 9 * DIMM, 12, 13, wO, wOs, DIMM, DIMM, T1);
    add_ln<<<NTOK, 256, 0, stream>>>(T1, X, ln1_g + (size_t)l * DIMM,
                                     ln1_b + (size_t)l * DIMM, X, 1e-5f);

    // FF: kan f1 -> relu -> kan f2 -> residual LN2
    kan_prep<<<NTOK * DIMM / LB, LB, 0, stream>>>(X, AP, DIMM, 9 * DIMM, NTOK);
    gemm(AP, 9 * DIMM, 14, 15, wF, wFs, DIMM, FFDIM, T0);
    relu_k<<<NTOK * FFDIM / LB, LB, 0, stream>>>(T0, NTOK * FFDIM);
    kan_prep<<<NTOK * FFDIM / LB, LB, 0, stream>>>(T0, AP, FFDIM, 9 * FFDIM, NTOK);
    gemm(AP, 9 * FFDIM, 16, 17, wF, wFs, FFDIM, DIMM, T1);
    add_ln<<<NTOK, 256, 0, stream>>>(T1, X, ln2_g + (size_t)l * DIMM,
                                     ln2_b + (size_t)l * DIMM, X, 1e-5f);
  }

  // Head: KAN -> gelu -> LN(no affine, eps 1e-12) -> KAN to vocab
  kan_prep<<<NTOK * DIMM / LB, LB, 0, stream>>>(X, AP, DIMM, 9 * DIMM, NTOK);
  gemm(AP, 9 * DIMM, 22, 23, 0, 0, DIMM, DIMM, T1);
  gelu_k<<<NTOK * DIMM / LB, LB, 0, stream>>>(T1, NTOK * DIMM);
  add_ln<<<NTOK, 256, 0, stream>>>(T1, nullptr, nullptr, nullptr, T1, 1e-12f);
  kan_prep<<<NTOK * DIMM / LB, LB, 0, stream>>>(T1, AP, DIMM, 9 * DIMM, NTOK);
  gemm(AP, 9 * DIMM, 24, 25, 0, 0, DIMM, VOCABN, out);
}
